// MultiHeadAttention_2284922602258
// MI455X (gfx1250) — matrix operand order repaired and checked by reading, not compiled, not run on hardware
//
#include <hip/hip_runtime.h>
#include <stdint.h>

// ---------------- problem constants ----------------
static constexpr int BATCH = 2;
static constexpr int SEQ   = 2048;
static constexpr int DMOD  = 1024;
static constexpr int NH    = 16;
static constexpr int DH    = 64;          // DMOD / NH
static constexpr int MTOT  = BATCH * SEQ; // 4096

// ---------------- WMMA types ----------------
typedef __attribute__((ext_vector_type(16))) __bf16         v16bf;
typedef __attribute__((ext_vector_type(8)))  float          v8f;
typedef __attribute__((ext_vector_type(8)))  unsigned short v8u16;
typedef int v4i __attribute__((vector_size(16)));

#define ASPACE(n) __attribute__((address_space(n)))

union FragU {
    struct { v8u16 lo, hi; } p;
    v16bf f;
};

static __device__ __forceinline__ v8f vzero8() {
    v8f z;
#pragma unroll
    for (int i = 0; i < 8; ++i) z[i] = 0.0f;
    return z;
}

static __device__ __forceinline__ unsigned short f2bf(float x) {
    unsigned int u = __builtin_bit_cast(unsigned int, x);
    u += 0x7FFFu + ((u >> 16) & 1u);   // round to nearest even
    return (unsigned short)(u >> 16);
}

// A-matrix fragment (16x32 bf16, MxK). Lane r=lane&15 holds row M=r.
// half h = lane>>4: VGPRs 0-3 hold K = h*8..h*8+7, VGPRs 4-7 hold K = 16+h*8..
static __device__ __forceinline__ v16bf load_a_frag(const unsigned short* base, int ld) {
    const int lane = threadIdx.x & 31;
    const int r = lane & 15, h = lane >> 4;
    const unsigned short* p = base + (size_t)r * ld + h * 8;
    FragU u;
    u.p.lo = *(const v8u16*)(p);
    u.p.hi = *(const v8u16*)(p + 16);
    return u.f;
}

// B-matrix fragment (32x16 bf16, KxN) from B^T stored row-major (row n holds K
// contiguously). Lane n=lane&15 = column; half h: K = h*8..h*8+7, 16+h*8..16+h*8+7.
static __device__ __forceinline__ v16bf load_b_frag(const unsigned short* base, int ld) {
    const int lane = threadIdx.x & 31;
    const int n = lane & 15, h = lane >> 4;
    const unsigned short* p = base + (size_t)n * ld + h * 8;
    FragU u;
    u.p.lo = *(const v8u16*)(p);
    u.p.hi = *(const v8u16*)(p + 16);
    return u.f;
}

static __device__ __forceinline__ v8f wmma_bf16(v16bf a, v16bf b, v8f c) {
    return __builtin_amdgcn_wmma_f32_16x16x32_bf16(false, a, false, b, (short)0, c, false, false);
}

// ---- async global->LDS copy of one 16B chunk (CDNA5 ASYNCcnt path) ----
static __device__ __forceinline__ void async_copy16(const unsigned short* g, unsigned short* l) {
#if __has_builtin(__builtin_amdgcn_global_load_async_to_lds_b128)
    __builtin_amdgcn_global_load_async_to_lds_b128(
        (ASPACE(1) v4i*)(size_t)g,
        (ASPACE(3) v4i*)(unsigned int)(size_t)l,
        0, 0);
#else
    *(v8u16*)l = *(const v8u16*)g;   // register-relay fallback
#endif
}

template <int N>
static __device__ __forceinline__ void wait_async() {
#if __has_builtin(__builtin_amdgcn_global_load_async_to_lds_b128)
#if __has_builtin(__builtin_amdgcn_s_wait_asynccnt)
    __builtin_amdgcn_s_wait_asynccnt((unsigned short)N);
#else
    if constexpr (N == 0) asm volatile("s_wait_asynccnt 0x0" ::: "memory");
    else                  asm volatile("s_wait_asynccnt 0x2" ::: "memory");
#endif
#endif
}

// Stage B-strip tile [128 cols x 32 K] bf16 into LDS buffer `buf`.
// 512 chunks of 16B; 256 threads x 2 async ops.
static __device__ __forceinline__ void stage_b(unsigned short (*Bs)[128][40],
                                               int buf, const unsigned short* W,
                                               int n0, int k) {
    const int tid = threadIdx.x;
#pragma unroll
    for (int i = 0; i < 2; ++i) {
        const int c = tid + i * 256;           // 0..511
        const int n = c >> 2, part = c & 3;    // 4 chunks per 32-K row
        const unsigned short* g = W + (size_t)(n0 + n) * DMOD + k + part * 8;
        async_copy16(g, &Bs[buf][n][part * 8]);
    }
}

// B fragment from LDS-staged tile (row-padded to 40 shorts = 80B).
static __device__ __forceinline__ v16bf load_b_frag_lds(const unsigned short* rowbase) {
    const int lane = threadIdx.x & 31;
    const int n = lane & 15, h = lane >> 4;
    const unsigned short* p = rowbase + n * 40 + h * 8;
    FragU u;
    u.p.lo = *(const v8u16*)(p);
    u.p.hi = *(const v8u16*)(p + 16);
    return u.f;
}

// ---------------- kernel 1: f32 -> bf16 cast ----------------
__global__ void __launch_bounds__(256)
cast_f32_bf16(const float* __restrict__ in, unsigned short* __restrict__ out, int n) {
    int i = blockIdx.x * 256 + threadIdx.x;
    if (i < n) out[i] = f2bf(in[i]);
}

// ---------------- kernel 2: fused Q/K/V projection ----------------
// y = x @ W^T. Block: 8 waves, 128 rows x 128-col strip; B staged in LDS
// (double-buffered async), each wave 16x128 (8 WMMAs / K-step per wave).
// sel 0 -> Q[B,H,S,dh], sel 1 -> K[B,H,S,dh], sel 2 -> V^T[B,H,dh,S]
__global__ void __launch_bounds__(256)
proj_qkv(const unsigned short* __restrict__ xb,
         const unsigned short* __restrict__ Wqb,
         const unsigned short* __restrict__ Wkb,
         const unsigned short* __restrict__ Wvb,
         unsigned short* __restrict__ Qb,
         unsigned short* __restrict__ Kb,
         unsigned short* __restrict__ Vt) {
    __shared__ unsigned short Bs[2][128][40];

    const int sel  = blockIdx.x >> 8;            // / (32*8)
    const int rem  = blockIdx.x & 255;
    const int mb   = rem >> 3;
    const int strip = rem & 7;
    const int wslot = threadIdx.x >> 5;
    const int m0   = mb * 128 + wslot * 16;
    const int n0   = strip * 128;
    const unsigned short* W = (sel == 0) ? Wqb : ((sel == 1) ? Wkb : Wvb);
    const unsigned short* Aptr = xb + (size_t)m0 * DMOD;

    v8f acc[8];
#pragma unroll
    for (int t = 0; t < 8; ++t) acc[t] = vzero8();

    stage_b(Bs, 0, W, n0, 0);
    for (int k = 0; k < DMOD; k += 32) {
        const int cur = (k >> 5) & 1;
        if (k + 32 < DMOD) { stage_b(Bs, cur ^ 1, W, n0, k + 32); wait_async<2>(); }
        else               { wait_async<0>(); }
        __syncthreads();
        __builtin_prefetch(Aptr + k + 256, 0, 1);
        v16bf a = load_a_frag(Aptr + k, DMOD);
#pragma unroll
        for (int t = 0; t < 8; ++t) {
            v16bf b = load_b_frag_lds(&Bs[cur][t * 16][0]);
            acc[t] = wmma_bf16(a, b, acc[t]);
        }
        __syncthreads();
    }

    const int lane = threadIdx.x & 31;
    const int col = lane & 15, half = lane >> 4;
#pragma unroll
    for (int t = 0; t < 8; ++t) {
#pragma unroll
        for (int v = 0; v < 8; ++v) {
            const int m  = m0 + v + 8 * half;
            const int b_ = m >> 11;              // / SEQ
            const int s_ = m & (SEQ - 1);
            const int n  = n0 + t * 16 + col;
            const int hh = n >> 6;               // head
            const int dl = n & (DH - 1);
            const unsigned short val = f2bf(acc[t][v]);
            if (sel == 2) {
                Vt[(((size_t)b_ * NH + hh) * DH + dl) * SEQ + s_] = val;
            } else if (sel == 1) {
                Kb[(((size_t)b_ * NH + hh) * SEQ + s_) * DH + dl] = val;
            } else {
                Qb[(((size_t)b_ * NH + hh) * SEQ + s_) * DH + dl] = val;
            }
        }
    }
}

// ---------------- kernel 3: causal flash attention ----------------
// One wave = one 16-row Q tile of one (b,h). Online softmax over 32-key chunks.
__global__ void __launch_bounds__(256)
flash_attn(const unsigned short* __restrict__ Qb,
           const unsigned short* __restrict__ Kb,
           const unsigned short* __restrict__ Vt,
           unsigned short* __restrict__ Ctx) {
    __shared__ unsigned short plds[8][16][40];   // per-wave P-transpose slab

    const int wslot = threadIdx.x >> 5;
    const int wave  = blockIdx.x * 8 + wslot;
    const int qtiles = SEQ / 16;                 // 128
    const int qt = wave % qtiles;
    const int hh = (wave / qtiles) % NH;
    const int b_ = wave / (qtiles * NH);
    const int q0 = qt * 16;

    const int lane = threadIdx.x & 31;
    const int col = lane & 15, half = lane >> 4;

    const unsigned short* Qbase = Qb + (((size_t)b_ * NH + hh) * SEQ + q0) * DH;
    const unsigned short* Kbase = Kb + (((size_t)b_ * NH + hh) * SEQ) * DH;
    const unsigned short* Vbase = Vt + (((size_t)b_ * NH + hh) * DH) * SEQ;

    const v16bf qa0 = load_a_frag(Qbase + 0,  DH);
    const v16bf qa1 = load_a_frag(Qbase + 32, DH);

    // NOTE: reference (faithfully buggy) scale is SEQ**-0.5, not dh**-0.5.
    const float scale = 0.022097086912079608f;   // 2048^-0.5

    float mrow[8], lrow[8];
    v8f ctxacc[4];
#pragma unroll
    for (int v = 0; v < 8; ++v) { mrow[v] = -3.0e30f; lrow[v] = 0.0f; }
#pragma unroll
    for (int t = 0; t < 4; ++t) ctxacc[t] = vzero8();

    for (int kc = 0; kc < q0 + 16; kc += 32) {
        // ---- scores: two 16-key tiles, 2 WMMAs each (K over dh=64) ----
        v8f st[2];
#pragma unroll
        for (int t = 0; t < 2; ++t) {
            const int kb = kc + t * 16;
            v16bf b0 = load_b_frag(Kbase + (size_t)kb * DH + 0,  DH);
            v16bf b1 = load_b_frag(Kbase + (size_t)kb * DH + 32, DH);
            v8f s = vzero8();
            s = wmma_bf16(qa0, b0, s);
            s = wmma_bf16(qa1, b1, s);
#pragma unroll
            for (int v = 0; v < 8; ++v) {
                const int row = q0 + v + 8 * half;
                const int key = kb + col;
                const float x = s[v] * scale;
                s[v] = (key > row) ? -3.0e30f : x;   // causal mask
            }
            st[t] = s;
        }

        // ---- online softmax: row reductions across the 16-lane half ----
        float corr[8];
#pragma unroll
        for (int v = 0; v < 8; ++v) {
            float mx = fmaxf(st[0][v], st[1][v]);
            mx = fmaxf(mx, __shfl_xor(mx, 1));
            mx = fmaxf(mx, __shfl_xor(mx, 2));
            mx = fmaxf(mx, __shfl_xor(mx, 4));
            mx = fmaxf(mx, __shfl_xor(mx, 8));
            const float mnew = fmaxf(mrow[v], mx);
            corr[v] = __expf(mrow[v] - mnew);
            const float p0 = __expf(st[0][v] - mnew);
            const float p1 = __expf(st[1][v] - mnew);
            st[0][v] = p0;
            st[1][v] = p1;
            float sum = p0 + p1;
            sum += __shfl_xor(sum, 1);
            sum += __shfl_xor(sum, 2);
            sum += __shfl_xor(sum, 4);
            sum += __shfl_xor(sum, 8);
            lrow[v] = lrow[v] * corr[v] + sum;
            mrow[v] = mnew;
        }

        // ---- transpose P (C-layout f32) -> A-fragment (bf16) via LDS ----
#pragma unroll
        for (int t = 0; t < 2; ++t)
#pragma unroll
            for (int v = 0; v < 8; ++v)
                plds[wslot][v + 8 * half][t * 16 + col] = f2bf(st[t][v]);
        asm volatile("s_wait_dscnt 0x0" ::: "memory");
        FragU pu;
        pu.p.lo = *(const v8u16*)(&plds[wslot][col][half * 8]);
        pu.p.hi = *(const v8u16*)(&plds[wslot][col][16 + half * 8]);
        const v16bf pa = pu.f;

        // ---- rescale running context, then ctx += P @ V (4 WMMAs) ----
#pragma unroll
        for (int t4 = 0; t4 < 4; ++t4)
#pragma unroll
            for (int v = 0; v < 8; ++v)
                ctxacc[t4][v] *= corr[v];
#pragma unroll
        for (int t4 = 0; t4 < 4; ++t4) {
            v16bf vb = load_b_frag(Vbase + (size_t)(t4 * 16) * SEQ + kc, SEQ);
            ctxacc[t4] = wmma_bf16(pa, vb, ctxacc[t4]);
        }
    }

    // ---- epilogue: normalize and store ctx as bf16 [B, S, H*dh] ----
#pragma unroll
    for (int t4 = 0; t4 < 4; ++t4) {
#pragma unroll
        for (int v = 0; v < 8; ++v) {
            const int s_ = q0 + v + 8 * half;
            const int dl = t4 * 16 + col;
            const float val = ctxacc[t4][v] / lrow[v];
            Ctx[((size_t)b_ * SEQ + s_) * DMOD + hh * DH + dl] = f2bf(val);
        }
    }
}

// ---------------- kernel 4: output projection + bias ----------------
// Same LDS-staged structure as proj_qkv; writes f32 to d_out.
__global__ void __launch_bounds__(256)
out_proj(const unsigned short* __restrict__ Ctx,
         const unsigned short* __restrict__ Wob,
         const float* __restrict__ bo,
         float* __restrict__ out) {
    __shared__ unsigned short Bs[2][128][40];

    const int mb    = blockIdx.x >> 3;
    const int strip = blockIdx.x & 7;
    const int wslot = threadIdx.x >> 5;
    const int m0    = mb * 128 + wslot * 16;
    const int n0    = strip * 128;
    const unsigned short* Aptr = Ctx + (size_t)m0 * DMOD;

    v8f acc[8];
#pragma unroll
    for (int t = 0; t < 8; ++t) acc[t] = vzero8();

    stage_b(Bs, 0, Wob, n0, 0);
    for (int k = 0; k < DMOD; k += 32) {
        const int cur = (k >> 5) & 1;
        if (k + 32 < DMOD) { stage_b(Bs, cur ^ 1, Wob, n0, k + 32); wait_async<2>(); }
        else               { wait_async<0>(); }
        __syncthreads();
        __builtin_prefetch(Aptr + k + 256, 0, 1);
        v16bf a = load_a_frag(Aptr + k, DMOD);
#pragma unroll
        for (int t = 0; t < 8; ++t) {
            v16bf b = load_b_frag_lds(&Bs[cur][t * 16][0]);
            acc[t] = wmma_bf16(a, b, acc[t]);
        }
        __syncthreads();
    }

    const int lane = threadIdx.x & 31;
    const int col = lane & 15, half = lane >> 4;
#pragma unroll
    for (int t = 0; t < 8; ++t) {
        const int n = n0 + t * 16 + col;
        const float bias = bo[n];
#pragma unroll
        for (int v = 0; v < 8; ++v) {
            const int m = m0 + v + 8 * half;
            out[(size_t)m * DMOD + n] = acc[t][v] + bias;
        }
    }
}

// ---------------- host launcher ----------------
extern "C" void kernel_launch(void* const* d_in, const int* in_sizes, int n_in,
                              void* d_out, int out_size, void* d_ws, size_t ws_size,
                              hipStream_t stream) {
    (void)in_sizes; (void)n_in; (void)out_size; (void)ws_size;

    const float* x  = (const float*)d_in[0];
    const float* Wq = (const float*)d_in[1];
    const float* Wk = (const float*)d_in[2];
    const float* Wv = (const float*)d_in[3];
    const float* Wo = (const float*)d_in[4];
    const float* bo = (const float*)d_in[5];
    float* out = (float*)d_out;

    char* ws = (char*)d_ws;
    size_t off = 0;
    auto alloc = [&](size_t bytes) -> void* {
        void* p = ws + off;
        off = (off + bytes + 255) & ~(size_t)255;
        return p;
    };

    const size_t NX = (size_t)MTOT * DMOD;          // 4,194,304
    const size_t NW = (size_t)DMOD * DMOD;          // 1,048,576
    const size_t NQ = (size_t)BATCH * NH * SEQ * DH;

    unsigned short* xb  = (unsigned short*)alloc(NX * 2);
    unsigned short* Wqb = (unsigned short*)alloc(NW * 2);
    unsigned short* Wkb = (unsigned short*)alloc(NW * 2);
    unsigned short* Wvb = (unsigned short*)alloc(NW * 2);
    unsigned short* Wob = (unsigned short*)alloc(NW * 2);
    unsigned short* Qb  = (unsigned short*)alloc(NQ * 2);
    unsigned short* Kb  = (unsigned short*)alloc(NQ * 2);
    unsigned short* Vt  = (unsigned short*)alloc(NQ * 2);
    unsigned short* Ctx = (unsigned short*)alloc(NX * 2);

    // 1) casts
    cast_f32_bf16<<<(int)((NX + 255) / 256), 256, 0, stream>>>(x,  xb,  (int)NX);
    cast_f32_bf16<<<(int)((NW + 255) / 256), 256, 0, stream>>>(Wq, Wqb, (int)NW);
    cast_f32_bf16<<<(int)((NW + 255) / 256), 256, 0, stream>>>(Wk, Wkb, (int)NW);
    cast_f32_bf16<<<(int)((NW + 255) / 256), 256, 0, stream>>>(Wv, Wvb, (int)NW);
    cast_f32_bf16<<<(int)((NW + 255) / 256), 256, 0, stream>>>(Wo, Wob, (int)NW);

    // 2) Q/K/V projections: 3 sel x 32 m-blocks x 8 strips = 768 blocks
    proj_qkv<<<768, 256, 0, stream>>>(xb, Wqb, Wkb, Wvb, Qb, Kb, Vt);

    // 3) flash attention: 2*16*128 = 4096 waves / 8 = 512 blocks
    flash_attn<<<512, 256, 0, stream>>>(Qb, Kb, Vt, Ctx);

    // 4) output projection: 32 m-blocks x 8 strips = 256 blocks
    out_proj<<<256, 256, 0, stream>>>(Ctx, Wob, bo, out);
}
